// DendriticAttentionNeuron_14405320311294
// MI455X (gfx1250) — compile-verified, matrix operand order checked by reading, not run on hardware
//
#include <hip/hip_runtime.h>
#include <cstdint>
#include <cstddef>

// ---------------------------------------------------------------------------
// DendriticAttentionNeuron fused forward for MI455X (gfx1250, wave32, WMMA).
//
// Shapes: B=8192, IN=4096, H=16, HD=64, HID=1024.
// Pipeline:
//   k_cvt_x      : x fp32 -> bf16                               (ws: xb)
//   k_transpose  : W[*][IN][HD] fp32 -> bf16 [HD][IN] (N-major) (ws: wbT)
//                  Wout[HID][IN]  fp32 -> bf16 [IN][HID]        (ws: woutT)
//   k_proj       : 7 fused GEMMs + gates + tanh/relu + q.k score (ws: vcomb, scores)
//   k_attn       : softmax over heads (temperature) scales v in place
//   k_out        : combined @ Wout + spike/v_new epilogue -> d_out
//
// GEMM staging uses CDNA5 GLOBAL_LOAD_ASYNC_TO_LDS_B128 (ASYNCcnt) with two
// LDS buffers when the builtins exist; otherwise falls back to synchronous
// VGPR staging (round-1 path).
//
// Workspace layout (bytes), assumed ws_size >= ~145 MB:
//   xb     @ 0          : 8192*4096*2    = 67,108,864
//   wbT    @ 67108864   : 7*16*64*4096*2 = 58,720,256
//   woutT  @ 125829120  : 4096*1024*2    = 8,388,608
//   vcomb  @ 134217728  : 8192*1024*2    = 16,777,216
//   scores @ 150994944  : 8192*16*4      = 524,288
// ---------------------------------------------------------------------------

typedef unsigned short u16;
typedef __attribute__((ext_vector_type(16))) __bf16 v16bf;
typedef __attribute__((ext_vector_type(8)))  float  v8f;
typedef __attribute__((ext_vector_type(4)))  int    v4i;

union FragBF { v16bf v; uint4 q[2]; };

#if __has_builtin(__builtin_amdgcn_global_load_async_to_lds_b128) && \
    __has_builtin(__builtin_amdgcn_s_wait_asynccnt)
#define USE_ASYNC_LDS 1
#else
#define USE_ASYNC_LDS 0
#endif

#if USE_ASYNC_LDS
// Builtin signature (from hipcc diagnostic): (v4i AS1*, v4i AS3*, imm, imm).
typedef __attribute__((address_space(1))) v4i gv4i;   // global int4
typedef __attribute__((address_space(3))) v4i sv4i;   // LDS int4

// 16B async copy: global (AS1) -> LDS (AS3), tracked by ASYNCcnt.
__device__ __forceinline__ void async_cp16(const u16* g, u16* l) {
    __builtin_amdgcn_global_load_async_to_lds_b128(
        (gv4i*)(g), (sv4i*)(l), 0, 0);
}
#endif

__device__ __forceinline__ u16 f2bf(float f) {
    union { float f; unsigned u; } v; v.f = f;
    unsigned r = v.u + 0x7FFFu + ((v.u >> 16) & 1u);   // round-to-nearest-even
    return (u16)(r >> 16);
}
__device__ __forceinline__ float bf2f(u16 b) {
    union { unsigned u; float f; } v; v.u = ((unsigned)b) << 16;
    return v.f;
}
__device__ __forceinline__ float sigmoidf_(float x) {
    return 1.0f / (1.0f + __expf(-x));
}

// ---------------------------------------------------------------------------
// Kernel 1: x fp32 -> bf16, 4 elements per thread.
// ---------------------------------------------------------------------------
__global__ __launch_bounds__(256) void k_cvt_x(const float* __restrict__ src,
                                               u16* __restrict__ dst) {
    size_t i = ((size_t)blockIdx.x * 256 + threadIdx.x) * 4;
    float4 v = *(const float4*)(src + i);
    struct __align__(8) U4 { u16 a, b, c, d; } o;
    o.a = f2bf(v.x); o.b = f2bf(v.y); o.c = f2bf(v.z); o.d = f2bf(v.w);
    *(U4*)(dst + i) = o;
}

// ---------------------------------------------------------------------------
// Kernel 2: tiled transpose + convert.  src: nmat matrices [R][C] fp32.
// dst: nmat matrices [C][R] bf16.  Grid: (R/64, C/64, nmat), 256 threads.
// ---------------------------------------------------------------------------
__global__ __launch_bounds__(256) void k_transpose(const float* __restrict__ src,
                                                   u16* __restrict__ dst,
                                                   int R, int C) {
    __shared__ float t[64][65];
    const size_t mOff = (size_t)blockIdx.z * (size_t)R * (size_t)C;
    src += mOff; dst += mOff;
    const int rt = blockIdx.x * 64, ct = blockIdx.y * 64;
    {
        int c = threadIdx.x & 63, r4 = threadIdx.x >> 6;
        #pragma unroll
        for (int i = 0; i < 16; i++) {
            int r = r4 + i * 4;
            t[r][c] = src[(size_t)(rt + r) * C + (ct + c)];
        }
    }
    __syncthreads();
    {
        int r = threadIdx.x & 63, c4 = threadIdx.x >> 6;
        #pragma unroll
        for (int i = 0; i < 16; i++) {
            int c = c4 + i * 4;
            dst[(size_t)(ct + c) * R + (rt + r)] = f2bf(t[r][c]);
        }
    }
}

// ---------------------------------------------------------------------------
// Kernel 3: fused projections.
// Block = (64 batch rows) x (1 head), 256 threads = 8 waves.
// Waves 0..6 each own one projection (qff,qg,qc,kff,kg,kc,vff) and compute a
// 64x64 bf16 WMMA tile (4x4 of 16x16x32 accumulators).  All waves stage.
// Double-buffered LDS, async global->LDS staging when available.
// ---------------------------------------------------------------------------
__global__ __launch_bounds__(256) void k_proj(const u16* __restrict__ xb,
                                              const u16* __restrict__ wbT,
                                              const float* __restrict__ bqg,
                                              const float* __restrict__ bkg,
                                              float* __restrict__ scores,
                                              u16* __restrict__ vcomb) {
    // two staging buffers of 32KB each: [A 64x32 | B 7x64x32] bf16
    // epilogue reuse: 7 x 32 x 64 fp32 = 56KB
    __shared__ __align__(16) unsigned char smem[65536];
    float* ep = (float*)smem;               // [7][32][64] (after barrier reuse)

    const int tid  = threadIdx.x;
    const int wid  = tid >> 5;
    const int lane = tid & 31;
    const int m0   = blockIdx.x * 64;
    const int h    = blockIdx.y;

    v8f acc[4][4];
    #pragma unroll
    for (int mi = 0; mi < 4; mi++)
        #pragma unroll
        for (int ni = 0; ni < 4; ni++)
            #pragma unroll
            for (int j = 0; j < 8; j++) acc[mi][ni][j] = 0.0f;

    // staging assignment: thread -> (row r, 16B chunk ch)
    const int r  = tid >> 2;
    const int ch = tid & 3;
    const u16* aSrc = xb + (size_t)(m0 + r) * 4096 + ch * 8;
    const u16* bSrc[7];
    #pragma unroll
    for (int p = 0; p < 7; p++)
        bSrc[p] = wbT + ((size_t)(p * 16 + h) * 64 + r) * 4096 + ch * 8;

    const int half = lane >> 4;   // which 16-lane group
    const int ln   = lane & 15;
    const int stOff = r * 32 + ch * 8;      // element offset within a tile

#if USE_ASYNC_LDS
    // ---- async double-buffered pipeline ----
    {   // prologue: tile 0 -> buffer 0
        u16* At = (u16*)(smem);
        u16* Bt = (u16*)(smem + 4096);
        async_cp16(aSrc, At + stOff);
        #pragma unroll
        for (int p = 0; p < 7; p++)
            async_cp16(bSrc[p], Bt + p * 2048 + stOff);
    }
    for (int k0 = 0; k0 < 4096; k0 += 32) {
        const int cur = (k0 >> 5) & 1;
        if (k0 + 32 < 4096) {               // issue next tile into other buffer
            u16* At = (u16*)(smem + (cur ^ 1) * 32768);
            u16* Bt = At + 2048;
            async_cp16(aSrc + k0 + 32, At + stOff);
            #pragma unroll
            for (int p = 0; p < 7; p++)
                async_cp16(bSrc[p] + k0 + 32, Bt + p * 2048 + stOff);
            __builtin_amdgcn_s_wait_asynccnt(8);   // current tile retired
        } else {
            __builtin_amdgcn_s_wait_asynccnt(0);
        }
        __syncthreads();

        if (wid < 7) {
            const u16* At = (const u16*)(smem + cur * 32768);
            const u16* bp = At + 2048 + wid * 2048;
            FragBF a[4], b[4];
            #pragma unroll
            for (int mi = 0; mi < 4; mi++) {
                const u16* s = At + (mi * 16 + ln) * 32 + half * 8;
                a[mi].q[0] = *(const uint4*)(s);
                a[mi].q[1] = *(const uint4*)(s + 16);
            }
            #pragma unroll
            for (int ni = 0; ni < 4; ni++) {
                const u16* s = bp + (ni * 16 + ln) * 32 + half * 8;
                b[ni].q[0] = *(const uint4*)(s);
                b[ni].q[1] = *(const uint4*)(s + 16);
            }
            #pragma unroll
            for (int mi = 0; mi < 4; mi++)
                #pragma unroll
                for (int ni = 0; ni < 4; ni++)
                    acc[mi][ni] = __builtin_amdgcn_wmma_f32_16x16x32_bf16(
                        false, a[mi].v, false, b[ni].v,
                        (short)0, acc[mi][ni], false, false);
        }
        __syncthreads();   // protect buf[cur^1] before next-next issue
    }
#else
    // ---- synchronous fallback (round-1 path) ----
    u16* At = (u16*)smem;
    u16* Bt = (u16*)(smem + 4096);
    for (int k0 = 0; k0 < 4096; k0 += 32) {
        __syncthreads();
        *(uint4*)(At + stOff) = *(const uint4*)(aSrc + k0);
        #pragma unroll
        for (int p = 0; p < 7; p++)
            *(uint4*)(Bt + p * 2048 + stOff) = *(const uint4*)(bSrc[p] + k0);
        if (k0 + 32 < 4096) {
            __builtin_prefetch(aSrc + k0 + 32, 0, 1);
            __builtin_prefetch(bSrc[0] + k0 + 32, 0, 1);
            __builtin_prefetch(bSrc[3] + k0 + 32, 0, 1);
            __builtin_prefetch(bSrc[6] + k0 + 32, 0, 1);
        }
        __syncthreads();
        if (wid < 7) {
            const u16* bp = Bt + wid * 2048;
            FragBF a[4], b[4];
            #pragma unroll
            for (int mi = 0; mi < 4; mi++) {
                const u16* s = At + (mi * 16 + ln) * 32 + half * 8;
                a[mi].q[0] = *(const uint4*)(s);
                a[mi].q[1] = *(const uint4*)(s + 16);
            }
            #pragma unroll
            for (int ni = 0; ni < 4; ni++) {
                const u16* s = bp + (ni * 16 + ln) * 32 + half * 8;
                b[ni].q[0] = *(const uint4*)(s);
                b[ni].q[1] = *(const uint4*)(s + 16);
            }
            #pragma unroll
            for (int mi = 0; mi < 4; mi++)
                #pragma unroll
                for (int ni = 0; ni < 4; ni++)
                    acc[mi][ni] = __builtin_amdgcn_wmma_f32_16x16x32_bf16(
                        false, a[mi].v, false, b[ni].v,
                        (short)0, acc[mi][ni], false, false);
        }
    }
#endif

    // ---- fused epilogue, two 32-row halves through LDS ----
    const float invtau = exp2f(-(1.0f + 4.0f * (float)h / 15.0f)); // 1/tau_h
    #pragma unroll
    for (int hm = 0; hm < 2; hm++) {
        __syncthreads();                 // staging/epilogue LDS reuse fence
        if (wid < 7) {
            float* dst = ep + wid * 2048;     // [32][64]
            #pragma unroll
            for (int ml = 0; ml < 2; ml++)
                #pragma unroll
                for (int ni = 0; ni < 4; ni++)
                    #pragma unroll
                    for (int j = 0; j < 8; j++)
                        dst[(ml * 16 + half * 8 + j) * 64 + ni * 16 + ln] =
                            acc[hm * 2 + ml][ni][j];
        }
        __syncthreads();

        // 256 threads cover 32x64: thread -> row er (0..31), 8 columns
        const int er    = tid >> 3;
        const int dbase = (tid & 7) * 8;
        float s = 0.0f;
        #pragma unroll
        for (int dd = 0; dd < 8; dd++) {
            const int d   = dbase + dd;
            const int idx = er * 64 + d;
            const float qff = ep[0 * 2048 + idx];
            const float qg  = ep[1 * 2048 + idx];
            const float qc  = ep[2 * 2048 + idx];
            const float kff = ep[3 * 2048 + idx];
            const float kg  = ep[4 * 2048 + idx];
            const float kc  = ep[5 * 2048 + idx];
            const float vff = ep[6 * 2048 + idx];
            const float qv = tanhf(invtau * (qff + sigmoidf_(qg + bqg[h * 64 + d]) * qc));
            const float kv = tanhf(invtau * (kff + sigmoidf_(kg + bkg[h * 64 + d]) * kc));
            const float vv = fmaxf(invtau * vff, 0.0f);
            s += qv * kv;
            vcomb[(size_t)(m0 + hm * 32 + er) * 1024 + h * 64 + d] = f2bf(vv);
        }
        // reduce over the 8 lanes sharing a row (wave32 shuffles)
        s += __shfl_xor(s, 1);
        s += __shfl_xor(s, 2);
        s += __shfl_xor(s, 4);
        if ((tid & 7) == 0)
            scores[(size_t)(m0 + hm * 32 + er) * 16 + h] = s * 0.125f; // /sqrt(64)
    }
}

// ---------------------------------------------------------------------------
// Kernel 4: softmax over heads (with temperature), scale v rows in place.
// ---------------------------------------------------------------------------
__global__ __launch_bounds__(256) void k_attn(const float* __restrict__ scores,
                                              const float* __restrict__ tempP,
                                              u16* __restrict__ vcomb) {
    const int b = blockIdx.x * 256 + threadIdx.x;
    const float invT = 1.0f / tempP[0];
    float a[16];
    float m = -3.0e38f;
    #pragma unroll
    for (int hh = 0; hh < 16; hh++) {
        a[hh] = scores[(size_t)b * 16 + hh];
        m = fmaxf(m, a[hh]);
    }
    float sum = 0.0f;
    #pragma unroll
    for (int hh = 0; hh < 16; hh++) {
        a[hh] = __expf((a[hh] - m) * invT);
        sum += a[hh];
    }
    const float inv = 1.0f / sum;
    uint4* row = (uint4*)(vcomb + (size_t)b * 1024);
    for (int c = 0; c < 128; c++) {
        const float w = a[c >> 3] * inv;     // 8 bf16 per uint4, 64 per head
        uint4 q = row[c];
        unsigned* pu = (unsigned*)&q;
        #pragma unroll
        for (int e = 0; e < 4; e++) {
            const float f0 = bf2f((u16)(pu[e] & 0xFFFFu)) * w;
            const float f1 = bf2f((u16)(pu[e] >> 16)) * w;
            pu[e] = ((unsigned)f2bf(f1) << 16) | (unsigned)f2bf(f0);
        }
        row[c] = q;
    }
}

// ---------------------------------------------------------------------------
// Kernel 5: out = combined[8192x1024] @ Wout (woutT is [4096][1024] bf16),
// fused spike / v_new epilogue.  Block tile 128x256, 8 waves of 64x64.
// Double-buffered LDS, async staging when available.
// ---------------------------------------------------------------------------
__global__ __launch_bounds__(256) void k_out(const u16* __restrict__ comb,
                                             const u16* __restrict__ woutT,
                                             float* __restrict__ out) {
    // buffer = [A 128x32 (8KB) | B 256x32 (16KB)] bf16, x2 = 48KB
    __shared__ __align__(16) unsigned char smem[49152];
    const int tid  = threadIdx.x;
    const int wid  = tid >> 5;
    const int lane = tid & 31;
    const int m0 = blockIdx.x * 128;
    const int n0 = blockIdx.y * 256;
    const int wm = wid >> 2;      // 0..1
    const int wn = wid & 3;       // 0..3
    const int half = lane >> 4;
    const int ln   = lane & 15;

    v8f acc[4][4];
    #pragma unroll
    for (int mi = 0; mi < 4; mi++)
        #pragma unroll
        for (int ni = 0; ni < 4; ni++)
            #pragma unroll
            for (int j = 0; j < 8; j++) acc[mi][ni][j] = 0.0f;

    // staging: A has 512 16B chunks (2/thread), B has 1024 (4/thread)
    const int ra0 = (tid + 0 * 256) >> 2, ca0 = (tid + 0 * 256) & 3;
    const int ra1 = (tid + 1 * 256) >> 2, ca1 = (tid + 1 * 256) & 3;
    const u16* aS0 = comb + (size_t)(m0 + ra0) * 1024 + ca0 * 8;
    const u16* aS1 = comb + (size_t)(m0 + ra1) * 1024 + ca1 * 8;
    const int aO0 = ra0 * 32 + ca0 * 8, aO1 = ra1 * 32 + ca1 * 8;
    const u16* bS[4];
    int bO[4];
    #pragma unroll
    for (int i = 0; i < 4; i++) {
        int cidx = tid + i * 256;
        int rr = cidx >> 2, cc = cidx & 3;
        bS[i] = woutT + (size_t)(n0 + rr) * 1024 + cc * 8;
        bO[i] = rr * 32 + cc * 8;
    }

#if USE_ASYNC_LDS
    {   // prologue: tile 0 -> buffer 0
        u16* At = (u16*)smem;
        u16* Bt = (u16*)(smem + 8192);
        async_cp16(aS0, At + aO0);
        async_cp16(aS1, At + aO1);
        #pragma unroll
        for (int i = 0; i < 4; i++) async_cp16(bS[i], Bt + bO[i]);
    }
#endif

    for (int k0 = 0; k0 < 1024; k0 += 32) {
#if USE_ASYNC_LDS
        const int cur = (k0 >> 5) & 1;
        if (k0 + 32 < 1024) {
            u16* At = (u16*)(smem + (cur ^ 1) * 24576);
            u16* Bt = At + 4096;
            async_cp16(aS0 + k0 + 32, At + aO0);
            async_cp16(aS1 + k0 + 32, At + aO1);
            #pragma unroll
            for (int i = 0; i < 4; i++) async_cp16(bS[i] + k0 + 32, Bt + bO[i]);
            __builtin_amdgcn_s_wait_asynccnt(6);
        } else {
            __builtin_amdgcn_s_wait_asynccnt(0);
        }
        __syncthreads();
        const u16* At = (const u16*)(smem + cur * 24576);
        const u16* Bt = At + 4096;
#else
        u16* At = (u16*)smem;
        u16* Bt = (u16*)(smem + 8192);
        __syncthreads();
        *(uint4*)(At + aO0) = *(const uint4*)(aS0 + k0);
        *(uint4*)(At + aO1) = *(const uint4*)(aS1 + k0);
        #pragma unroll
        for (int i = 0; i < 4; i++)
            *(uint4*)(Bt + bO[i]) = *(const uint4*)(bS[i] + k0);
        __syncthreads();
#endif

        FragBF a[4], b[4];
        #pragma unroll
        for (int mi = 0; mi < 4; mi++) {
            const u16* s = At + (wm * 64 + mi * 16 + ln) * 32 + half * 8;
            a[mi].q[0] = *(const uint4*)(s);
            a[mi].q[1] = *(const uint4*)(s + 16);
        }
        #pragma unroll
        for (int ni = 0; ni < 4; ni++) {
            const u16* s = Bt + (wn * 64 + ni * 16 + ln) * 32 + half * 8;
            b[ni].q[0] = *(const uint4*)(s);
            b[ni].q[1] = *(const uint4*)(s + 16);
        }
        #pragma unroll
        for (int mi = 0; mi < 4; mi++)
            #pragma unroll
            for (int ni = 0; ni < 4; ni++)
                acc[mi][ni] = __builtin_amdgcn_wmma_f32_16x16x32_bf16(
                    false, a[mi].v, false, b[ni].v,
                    (short)0, acc[mi][ni], false, false);
#if USE_ASYNC_LDS
        __syncthreads();   // protect buf[cur^1] before next-next issue
#endif
    }

    // fused epilogue: v_soma = 0.5*out; spike = sigmoid(4*(vs-1)); vnew = vs - spike
    const size_t VOFF = (size_t)8192 * 4096;   // second output (v_new) offset
    #pragma unroll
    for (int mi = 0; mi < 4; mi++)
        #pragma unroll
        for (int ni = 0; ni < 4; ni++)
            #pragma unroll
            for (int j = 0; j < 8; j++) {
                const int row = m0 + wm * 64 + mi * 16 + half * 8 + j;
                const int col = n0 + wn * 64 + ni * 16 + ln;
                const float vs    = 0.5f * acc[mi][ni][j];
                const float spike = sigmoidf_(4.0f * (vs - 1.0f));
                out[(size_t)row * 4096 + col]        = spike;
                out[VOFF + (size_t)row * 4096 + col] = vs - spike;
            }
}

// ---------------------------------------------------------------------------
// Host-side launcher.
// ---------------------------------------------------------------------------
extern "C" void kernel_launch(void* const* d_in, const int* in_sizes, int n_in,
                              void* d_out, int out_size, void* d_ws, size_t ws_size,
                              hipStream_t stream) {
    (void)in_sizes; (void)n_in; (void)out_size; (void)ws_size;

    const float* x    = (const float*)d_in[0];
    const float* W[7] = {
        (const float*)d_in[1],  // Wq_ff
        (const float*)d_in[2],  // Wq_gate
        (const float*)d_in[4],  // Wq_ctx
        (const float*)d_in[5],  // Wk_ff
        (const float*)d_in[6],  // Wk_gate
        (const float*)d_in[8],  // Wk_ctx
        (const float*)d_in[9],  // Wv_ff
    };
    const float* bqg  = (const float*)d_in[3];
    const float* bkg  = (const float*)d_in[7];
    const float* Wout = (const float*)d_in[10];
    const float* temp = (const float*)d_in[11];

    char* ws = (char*)d_ws;
    u16*   xb     = (u16*)(ws);
    u16*   wbT    = (u16*)(ws + 67108864);
    u16*   woutT  = (u16*)(ws + 125829120);
    u16*   vcomb  = (u16*)(ws + 134217728);
    float* scores = (float*)(ws + 150994944);

    // 1) x -> bf16 (8192*4096 elems, 4/thread)
    k_cvt_x<<<32768, 256, 0, stream>>>(x, xb);

    // 2) weights -> bf16, transposed to [N][K]
    for (int p = 0; p < 7; p++)
        k_transpose<<<dim3(64, 1, 16), 256, 0, stream>>>(
            W[p], wbT + (size_t)p * 16 * 64 * 4096, 4096, 64);
    k_transpose<<<dim3(16, 64, 1), 256, 0, stream>>>(Wout, woutT, 1024, 4096);

    // 3) fused projections + gates + scores + v
    k_proj<<<dim3(128, 16), 256, 0, stream>>>(xb, wbT, bqg, bkg, scores, vcomb);

    // 4) softmax over heads, scale v
    k_attn<<<32, 256, 0, stream>>>(scores, temp, vcomb);

    // 5) output GEMM + spike / v_new
    k_out<<<dim3(64, 16), 256, 0, stream>>>(vcomb, woutT, (float*)d_out);
}
